// MoEAttention_50337016709687
// MI455X (gfx1250) — matrix operand order repaired and checked by reading, not compiled, not run on hardware
//
#include <hip/hip_runtime.h>
#include <hip/hip_bf16.h>
#include <stdint.h>

// ---------------------------------------------------------------------------
// MoE attention for MI455X (gfx1250, wave32, WMMA).
// All GEMMs: bf16 inputs, v_wmma_f32_16x16x32_bf16, fp32 accumulate.
// Expert stage folded to one per-batch GEMM (gating is per-sample).
// GEMM uses double-buffered global_load_async_to_lds_b128 staging (ASYNCcnt).
// ---------------------------------------------------------------------------

typedef __attribute__((ext_vector_type(16))) __bf16 v16bf;
typedef __attribute__((ext_vector_type(8)))  float  v8f;

struct alignas(16) U128 { unsigned int x[4]; };

union V16 {
  v16bf v;
  __bf16 h[16];
  unsigned int u[8];
};

#if defined(__has_builtin)
#if __has_builtin(__builtin_amdgcn_global_load_async_to_lds_b128)
#define HAVE_ASYNC_LDS 1
#endif
#endif

// builtin signature (from clang diagnostic): pointers to 4 x i32 vectors,
// global source in AS1 ("__device__"), LDS destination in AS3 ("__shared__")
typedef int v4i __attribute__((vector_size(16)));
typedef __attribute__((address_space(1))) v4i v4i_g;
typedef __attribute__((address_space(3))) v4i v4i_s;

__device__ __forceinline__ void async_wait0() {
#if defined(__has_builtin)
#if __has_builtin(__builtin_amdgcn_s_wait_asynccnt)
  __builtin_amdgcn_s_wait_asynccnt(0);
  return;
#else
  asm volatile("s_wait_asynccnt 0x0" ::: "memory");
  return;
#endif
#endif
}

__device__ __forceinline__ v8f wmma_bf16(v16bf a, v16bf b, v8f c) {
  // (neg_a, A, neg_b, B, c_mod, C, reuse_a, reuse_b)
  return __builtin_amdgcn_wmma_f32_16x16x32_bf16(false, a, false, b, (short)0, c,
                                                 false, false);
}

constexpr int cB = 4, cS = 1024, cD = 768, cH = 12, cDH = 64, cE = 4;

// --------------------------- fp32 -> bf16 convert ---------------------------
__global__ void cvt_f32_to_bf16(const float* __restrict__ in,
                                __bf16* __restrict__ out, int n) {
  int i = blockIdx.x * 256 + threadIdx.x;
  if (i < n) out[i] = (__bf16)in[i];
}

// ------------------------------- tiled GEMM ---------------------------------
// C[M,N] = A[M,K] * W[N,K]^T + bias[N]
// 128 threads = 4 waves; block tile 128x64; each wave: 32 rows x 64 cols
// mode 0: row-major out (f32 and/or bf16)
// mode 1: split-heads bf16   out[((b*H+h)*S+s)*DH+dh]
// mode 2: split-heads-T bf16 out[((b*H+h)*DH+dh)*S+s]   (for V)
// wBatchStride/biasBatchStride: per-batch weights (folded experts), batch=m/cS
__global__ __launch_bounds__(128) void gemm_bf16(
    const __bf16* __restrict__ A, const __bf16* __restrict__ W,
    const float* __restrict__ bias, float* __restrict__ outF,
    __bf16* __restrict__ outH, int M, int N, int Kd, int mode,
    int wBatchStride, int biasBatchStride) {
#if defined(HAVE_ASYNC_LDS)
  __shared__ __align__(16) __bf16 As[2][128][32];
  __shared__ __align__(16) __bf16 Ws[2][64][32];
#else
  __shared__ __align__(16) __bf16 As[1][128][32];
  __shared__ __align__(16) __bf16 Ws[1][64][32];
#endif
  const int tid = threadIdx.x;
  const int wid = tid >> 5;
  const int lane = tid & 31;
  const int g = lane >> 4;  // half-wave group
  const int ln = lane & 15;
  const int m0 = blockIdx.x * 128;
  const int n0 = blockIdx.y * 64;
  if (wBatchStride) W += (size_t)(m0 / cS) * wBatchStride;
  if (biasBatchStride) bias += (size_t)(m0 / cS) * biasBatchStride;

  // staging geometry: chunk = 16B; A: 4 chunks/thread, W: 2 chunks/thread
  const int srow = tid >> 2;        // 0..31
  const int scol = (tid & 3) * 8;   // 0,8,16,24

  v8f zero8 = {0.f, 0.f, 0.f, 0.f, 0.f, 0.f, 0.f, 0.f};
  v8f acc[2][4];
#pragma unroll
  for (int i = 0; i < 2; ++i)
#pragma unroll
    for (int j = 0; j < 4; ++j) acc[i][j] = zero8;

#if defined(HAVE_ASYNC_LDS)
  auto stage = [&](int buf, int k0) {
#pragma unroll
    for (int c = 0; c < 4; ++c) {
      __builtin_amdgcn_global_load_async_to_lds_b128(
          (v4i_g*)(void*)&A[(size_t)(m0 + srow + c * 32) * Kd + k0 + scol],
          (v4i_s*)(void*)&As[buf][srow + c * 32][scol], 0, 0);
    }
#pragma unroll
    for (int c = 0; c < 2; ++c) {
      __builtin_amdgcn_global_load_async_to_lds_b128(
          (v4i_g*)(void*)&W[(size_t)(n0 + srow + c * 32) * Kd + k0 + scol],
          (v4i_s*)(void*)&Ws[buf][srow + c * 32][scol], 0, 0);
    }
  };
  // prologue: fill buffer 0
  stage(0, 0);
  async_wait0();
  __syncthreads();
#endif

  int cur = 0;
  for (int k0 = 0; k0 < Kd; k0 += 32) {
#if defined(HAVE_ASYNC_LDS)
    const bool more = (k0 + 32 < Kd);
    if (more) stage(cur ^ 1, k0 + 32);  // async prefetch next tile, no wait yet
#else
    // fallback: all global loads, then all LDS stores (no per-chunk stall)
    U128 ta[4], tw[2];
#pragma unroll
    for (int c = 0; c < 4; ++c)
      ta[c] = *(const U128*)&A[(size_t)(m0 + srow + c * 32) * Kd + k0 + scol];
#pragma unroll
    for (int c = 0; c < 2; ++c)
      tw[c] = *(const U128*)&W[(size_t)(n0 + srow + c * 32) * Kd + k0 + scol];
#pragma unroll
    for (int c = 0; c < 4; ++c) *(U128*)&As[0][srow + c * 32][scol] = ta[c];
#pragma unroll
    for (int c = 0; c < 2; ++c) *(U128*)&Ws[0][srow + c * 32][scol] = tw[c];
    __syncthreads();
#endif

    // A fragments (rows wid*32 + i*16 + ln); per-lane data is two 16B runs
    V16 af[2];
#pragma unroll
    for (int i = 0; i < 2; ++i) {
      const int arow = wid * 32 + i * 16 + ln;
      *(U128*)&af[i].u[0] = *(const U128*)&As[cur][arow][8 * g];
      *(U128*)&af[i].u[4] = *(const U128*)&As[cur][arow][16 + 8 * g];
    }
#pragma unroll
    for (int j = 0; j < 4; ++j) {
      // B fragment: lane n = j*16+ln, two 16B runs (kk = 8*g.., 16+8*g..)
      V16 wf;
      const int wrow = j * 16 + ln;
      *(U128*)&wf.u[0] = *(const U128*)&Ws[cur][wrow][8 * g];
      *(U128*)&wf.u[4] = *(const U128*)&Ws[cur][wrow][16 + 8 * g];
#pragma unroll
      for (int i = 0; i < 2; ++i) acc[i][j] = wmma_bf16(af[i].v, wf.v, acc[i][j]);
    }

#if defined(HAVE_ASYNC_LDS)
    if (more) async_wait0();  // next tile landed
    __syncthreads();          // all waves done reading cur / writing cur^1
    cur ^= 1;
#else
    __syncthreads();
#endif
  }

  // epilogue: C element (r,lane) -> m = m0+wid*32+i*16+r+8g, n = n0+j*16+ln
#pragma unroll
  for (int j = 0; j < 4; ++j) {
    const int n = n0 + j * 16 + ln;
    const float bv = bias ? bias[n] : 0.f;
#pragma unroll
    for (int i = 0; i < 2; ++i) {
#pragma unroll
      for (int r = 0; r < 8; ++r) {
        const int m = m0 + wid * 32 + i * 16 + r + 8 * g;
        float val = acc[i][j][r] + bv;
        if (mode == 0) {
          if (outF) outF[(size_t)m * N + n] = val;
          if (outH) outH[(size_t)m * N + n] = (__bf16)val;
        } else {
          int b = m / cS, s = m % cS;
          int h = n / cDH, dh = n % cDH;
          size_t off = (mode == 1)
                           ? ((((size_t)b * cH + h) * cS + s) * cDH + dh)
                           : ((((size_t)b * cH + h) * cDH + dh) * cS + s);
          outH[off] = (__bf16)val;
        }
      }
    }
  }
}

// --------------------------- flash attention --------------------------------
// Q,K: [B*H, S, DH] bf16 ; Vt: [B*H, DH, S] bf16 ; ctx out: [B,S,D] bf16
// 128 threads = 4 waves; each wave: 16 query rows, online softmax over S keys
__global__ __launch_bounds__(128) void attn_fwd(const __bf16* __restrict__ Q,
                                                const __bf16* __restrict__ Kc,
                                                const __bf16* __restrict__ Vt,
                                                __bf16* __restrict__ ctx) {
  __shared__ __align__(16) __bf16 Plds[4][16][32];  // wave-private P tiles
  const int wid = threadIdx.x >> 5;
  const int lane = threadIdx.x & 31;
  const int g = lane >> 4, ln = lane & 15;
  const int bh = blockIdx.y;
  const int b = bh / cH, h = bh % cH;
  const int q0 = blockIdx.x * 64 + wid * 16;
  const __bf16* Qh = Q + (size_t)bh * cS * cDH;
  const __bf16* Kh = Kc + (size_t)bh * cS * cDH;
  const __bf16* Vh = Vt + (size_t)bh * cDH * cS;

  // Q fragments (dh 0..31 and 32..63), A-layout: two 16B runs each
  V16 aq0, aq1;
  {
    const size_t qro = (size_t)(q0 + ln) * cDH;
    *(U128*)&aq0.u[0] = *(const U128*)&Qh[qro + 8 * g];
    *(U128*)&aq0.u[4] = *(const U128*)&Qh[qro + 16 + 8 * g];
    *(U128*)&aq1.u[0] = *(const U128*)&Qh[qro + 32 + 8 * g];
    *(U128*)&aq1.u[4] = *(const U128*)&Qh[qro + 48 + 8 * g];
  }

  float m8[8], l8[8];
  v8f zero8 = {0.f, 0.f, 0.f, 0.f, 0.f, 0.f, 0.f, 0.f};
  v8f acc[4];
#pragma unroll
  for (int r = 0; r < 8; ++r) { m8[r] = -1e30f; l8[r] = 0.f; }
#pragma unroll
  for (int j = 0; j < 4; ++j) acc[j] = zero8;

  const float scale = 0.125f;  // 1/sqrt(64)

  for (int kb = 0; kb < cS; kb += 32) {
    v8f st[2];
    float tmax[8];
#pragma unroll
    for (int r = 0; r < 8; ++r) tmax[r] = -1e30f;

#pragma unroll
    for (int t = 0; t < 2; ++t) {
      // K fragments: B[kk=dh][n=key]; lane reads two 16B runs of dh values
      V16 k0f, k1f;
      const size_t kro = (size_t)(kb + t * 16 + ln) * cDH;
      *(U128*)&k0f.u[0] = *(const U128*)&Kh[kro + 8 * g];
      *(U128*)&k0f.u[4] = *(const U128*)&Kh[kro + 16 + 8 * g];
      *(U128*)&k1f.u[0] = *(const U128*)&Kh[kro + 32 + 8 * g];
      *(U128*)&k1f.u[4] = *(const U128*)&Kh[kro + 48 + 8 * g];
      v8f c = wmma_bf16(aq0.v, k0f.v, zero8);
      c = wmma_bf16(aq1.v, k1f.v, c);
#pragma unroll
      for (int r = 0; r < 8; ++r) {
        float v = c[r] * scale;
        c[r] = v;
#pragma unroll
        for (int o = 1; o < 16; o <<= 1) v = fmaxf(v, __shfl_xor(v, o, 32));
        tmax[r] = fmaxf(tmax[r], v);
      }
      st[t] = c;
    }

    float corr[8];
#pragma unroll
    for (int r = 0; r < 8; ++r) {
      float mn = fmaxf(m8[r], tmax[r]);
      corr[r] = __expf(m8[r] - mn);
      m8[r] = mn;
    }
#pragma unroll
    for (int j = 0; j < 4; ++j)
#pragma unroll
      for (int r = 0; r < 8; ++r) acc[j][r] *= corr[r];

    float rsum[8];
#pragma unroll
    for (int r = 0; r < 8; ++r) rsum[r] = 0.f;
#pragma unroll
    for (int t = 0; t < 2; ++t) {
#pragma unroll
      for (int r = 0; r < 8; ++r) {
        float p = __expf(st[t][r] - m8[r]);
        rsum[r] += p;
        // C-layout -> LDS [row][key]; same-wave DS ops are in-order
        Plds[wid][r + 8 * g][t * 16 + ln] = (__bf16)p;
      }
    }
#pragma unroll
    for (int r = 0; r < 8; ++r) {
      float v = rsum[r];
#pragma unroll
      for (int o = 1; o < 16; o <<= 1) v += __shfl_xor(v, o, 32);
      l8[r] = l8[r] * corr[r] + v;
    }

    // reload P in A-fragment layout (row = ln): two 16B runs
    V16 ap;
    *(U128*)&ap.u[0] = *(const U128*)&Plds[wid][ln][8 * g];
    *(U128*)&ap.u[4] = *(const U128*)&Plds[wid][ln][16 + 8 * g];

    // V fragments from transposed V: B[kk=key_local][n=dh], contiguous in s
#pragma unroll
    for (int j = 0; j < 4; ++j) {
      V16 vf;
      const size_t vro = (size_t)(j * 16 + ln) * cS + kb + 8 * g;
      *(U128*)&vf.u[0] = *(const U128*)&Vh[vro];
      *(U128*)&vf.u[4] = *(const U128*)&Vh[vro + 16];
      acc[j] = wmma_bf16(ap.v, vf.v, acc[j]);
    }
  }

  // write merged-head ctx [B,S,D]
#pragma unroll
  for (int j = 0; j < 4; ++j) {
    const int d = h * cDH + j * 16 + ln;
#pragma unroll
    for (int r = 0; r < 8; ++r) {
      const int s = q0 + r + 8 * g;
      ctx[((size_t)b * cS + s) * cD + d] = (__bf16)(acc[j][r] / l8[r]);
    }
  }
}

// -------------------------- gating: mean over S -----------------------------
__global__ void col_mean_kernel(const __bf16* __restrict__ ctx,
                                float* __restrict__ mean) {
  int i = blockIdx.x * 256 + threadIdx.x;
  if (i >= cB * cD) return;
  int b = i / cD, d = i % cD;
  const __bf16* p = ctx + (size_t)b * cS * cD + d;
  float sum = 0.f;
  for (int s = 0; s < cS; ++s) sum += (float)p[(size_t)s * cD];
  mean[i] = sum * (1.0f / cS);
}

// ----------------- gate logits + softmax + top-2 combine weights ------------
__global__ void gate_kernel(const float* __restrict__ mean,
                            const float* __restrict__ Wg,
                            const float* __restrict__ bg,
                            float* __restrict__ wout) {
  __shared__ float logits[cB][cE];
  int t = threadIdx.x;
  if (t < cB * cE) {
    int b = t / cE, e = t % cE;
    float s = bg[e];
    const float* mp = mean + b * cD;
    const float* wp = Wg + e * cD;
    for (int d = 0; d < cD; ++d) s += mp[d] * wp[d];
    logits[b][e] = s;
  }
  __syncthreads();
  if (t < cB) {
    float l[cE], mx = -1e30f;
    for (int e = 0; e < cE; ++e) { l[e] = logits[t][e]; mx = fmaxf(mx, l[e]); }
    float sum = 0.f;
    for (int e = 0; e < cE; ++e) { l[e] = __expf(l[e] - mx); sum += l[e]; }
    for (int e = 0; e < cE; ++e) l[e] /= sum;
    int i1 = 0;
    for (int e = 1; e < cE; ++e) if (l[e] > l[i1]) i1 = e;
    int i2 = (i1 == 0) ? 1 : 0;
    for (int e = 0; e < cE; ++e) if (e != i1 && l[e] > l[i2]) i2 = e;
    for (int e = 0; e < cE; ++e)
      wout[t * cE + e] = (e == i1 || e == i2) ? l[e] : 0.f;
  }
}

// ----------- fold experts: Weff[b] = sum_e w[b,e]*W_exp[e] (bf16) -----------
__global__ void fold_experts(const float* __restrict__ Wexp,
                             const float* __restrict__ bexp,
                             const float* __restrict__ w,
                             __bf16* __restrict__ Weff,
                             float* __restrict__ beff) {
  size_t i = (size_t)blockIdx.x * 256 + threadIdx.x;
  const size_t per = (size_t)cD * cD;
  if (i >= (size_t)cB * per) return;
  int b = (int)(i / per);
  size_t nk = i % per;
  float w0 = w[b * cE + 0], w1 = w[b * cE + 1];
  float w2 = w[b * cE + 2], w3 = w[b * cE + 3];
  float s = w0 * Wexp[nk] + w1 * Wexp[per + nk] + w2 * Wexp[2 * per + nk] +
            w3 * Wexp[3 * per + nk];
  Weff[i] = (__bf16)s;
  if (nk < (size_t)cD) {
    float bb = w0 * bexp[nk] + w1 * bexp[cD + nk] + w2 * bexp[2 * cD + nk] +
               w3 * bexp[3 * cD + nk];
    beff[(size_t)b * cD + nk] = bb;
  }
}

// ------------------------------- launcher -----------------------------------
extern "C" void kernel_launch(void* const* d_in, const int* in_sizes, int n_in,
                              void* d_out, int out_size, void* d_ws,
                              size_t ws_size, hipStream_t stream) {
  const float* x    = (const float*)d_in[0];
  const float* Wq   = (const float*)d_in[1];
  const float* bq   = (const float*)d_in[2];
  const float* Wk   = (const float*)d_in[3];
  const float* bk   = (const float*)d_in[4];
  const float* Wv   = (const float*)d_in[5];
  const float* bv   = (const float*)d_in[6];
  const float* Wexp = (const float*)d_in[7];
  const float* bexp = (const float*)d_in[8];
  const float* Wg   = (const float*)d_in[9];
  const float* bg   = (const float*)d_in[10];
  const float* Wo   = (const float*)d_in[11];
  const float* bo   = (const float*)d_in[12];
  float* out = (float*)d_out;

  char* ws = (char*)d_ws;
  size_t off = 0;
  auto alloc = [&](size_t bytes) -> void* {
    void* p = ws + off;
    off = (off + bytes + 255) & ~(size_t)255;
    return p;
  };

  const int nX = cB * cS * cD;  // 3,145,728
  const int nW = cD * cD;       // 589,824
  __bf16* xb   = (__bf16*)alloc((size_t)nX * 2);
  __bf16* wqb  = (__bf16*)alloc((size_t)nW * 2);
  __bf16* wkb  = (__bf16*)alloc((size_t)nW * 2);
  __bf16* wvb  = (__bf16*)alloc((size_t)nW * 2);
  __bf16* wob  = (__bf16*)alloc((size_t)nW * 2);
  __bf16* qb   = (__bf16*)alloc((size_t)nX * 2);
  __bf16* kbuf = (__bf16*)alloc((size_t)nX * 2);
  __bf16* vtb  = (__bf16*)alloc((size_t)nX * 2);
  __bf16* ctxb = (__bf16*)alloc((size_t)nX * 2);
  __bf16* moeb = (__bf16*)alloc((size_t)nX * 2);
  float*  mean = (float*)alloc((size_t)cB * cD * 4);
  float* wgate = (float*)alloc((size_t)cB * cE * 4);
  __bf16* weff = (__bf16*)alloc((size_t)cB * nW * 2);
  float*  beff = (float*)alloc((size_t)cB * cD * 4);

  // fp32 -> bf16
  cvt_f32_to_bf16<<<(nX + 255) / 256, 256, 0, stream>>>(x, xb, nX);
  cvt_f32_to_bf16<<<(nW + 255) / 256, 256, 0, stream>>>(Wq, wqb, nW);
  cvt_f32_to_bf16<<<(nW + 255) / 256, 256, 0, stream>>>(Wk, wkb, nW);
  cvt_f32_to_bf16<<<(nW + 255) / 256, 256, 0, stream>>>(Wv, wvb, nW);
  cvt_f32_to_bf16<<<(nW + 255) / 256, 256, 0, stream>>>(Wo, wob, nW);

  const int M = cB * cS;         // 4096
  dim3 gg(M / 128, cD / 64, 1);  // (32, 12)

  // QKV projections (split heads; V stored transposed)
  gemm_bf16<<<gg, 128, 0, stream>>>(xb, wqb, bq, nullptr, qb,   M, cD, cD, 1, 0, 0);
  gemm_bf16<<<gg, 128, 0, stream>>>(xb, wkb, bk, nullptr, kbuf, M, cD, cD, 1, 0, 0);
  gemm_bf16<<<gg, 128, 0, stream>>>(xb, wvb, bv, nullptr, vtb,  M, cD, cD, 2, 0, 0);

  // attention -> ctx [B,S,D] bf16
  dim3 ga(cS / 64, cB * cH, 1);  // (16, 48)
  attn_fwd<<<ga, 128, 0, stream>>>(qb, kbuf, vtb, ctxb);

  // gating
  col_mean_kernel<<<(cB * cD + 255) / 256, 256, 0, stream>>>(ctxb, mean);
  gate_kernel<<<1, 64, 0, stream>>>(mean, Wg, bg, wgate);

  // fold experts into per-batch weight/bias
  fold_experts<<<(int)(((size_t)cB * nW + 255) / 256), 256, 0, stream>>>(
      Wexp, bexp, wgate, weff, beff);

  // moe = ctx @ Weff[b]^T + beff[b]   (batched weights)
  gemm_bf16<<<gg, 128, 0, stream>>>(ctxb, weff, beff, nullptr, moeb, M, cD, cD,
                                    0, nW, cD);

  // final projection: out = moe @ Wo^T + bo  (fp32 out)
  gemm_bf16<<<gg, 128, 0, stream>>>(moeb, wob, bo, out, nullptr, M, cD, cD, 0,
                                    0, 0);
}
